// MwAN_27376121545063
// MI455X (gfx1250) — compile-verified, matrix operand order checked by reading, not run on hardware
//
#include <hip/hip_runtime.h>

// ---------------------------------------------------------------------------
// MwAN forward for MI455X (gfx1250, wave32).  All matmuls run on
// V_WMMA_F32_16X16X32_F16 (fp32 accumulate).  Weight-side (B) operands and
// GEMM-only activations are pre-converted to f16 once per call, so the hot
// loops issue pure b128 loads with no f32->f16 conversion; mixed-consumer
// activations stay fp32 and convert on load (amortized over 4-16 WMMAs).
// All GEMM dims are tile-exact (M%16, N%64, K%32 == 0): zero bounds checks.
// ---------------------------------------------------------------------------

typedef __attribute__((ext_vector_type(16))) _Float16 v16h;
typedef __attribute__((ext_vector_type(8)))  _Float16 v8h;
typedef __attribute__((ext_vector_type(8)))  float    v8f;

#define B_   16
#define LP_  128
#define LC_  128
#define H_   128
#define E_   300
#define EP_  320      // E padded to a multiple of 32
#define G3_  384      // 3*H
#define D2_  256      // 2*H
#define DA_  1536     // 12*H
#define SLD_ 129      // S-tile LDS row stride (odd -> conflict-free column reads)

// K index pattern for the 16-bit A/B fragment of V_WMMA_*_16x16x32 (wave32):
// lane group base kb = (lane>>4)*8; elements 0..7 -> K kb+0..7,
// elements 8..15 -> K kb+16..23.  I.e. two contiguous 8-element runs.
static __device__ __forceinline__ v16h cvt16(float4 f0, float4 f1, float4 f2, float4 f3)
{
    v16h r;
    r[0]  = (_Float16)f0.x; r[1]  = (_Float16)f0.y; r[2]  = (_Float16)f0.z; r[3]  = (_Float16)f0.w;
    r[4]  = (_Float16)f1.x; r[5]  = (_Float16)f1.y; r[6]  = (_Float16)f1.z; r[7]  = (_Float16)f1.w;
    r[8]  = (_Float16)f2.x; r[9]  = (_Float16)f2.y; r[10] = (_Float16)f2.z; r[11] = (_Float16)f2.w;
    r[12] = (_Float16)f3.x; r[13] = (_Float16)f3.y; r[14] = (_Float16)f3.z; r[15] = (_Float16)f3.w;
    return r;
}

// fp32 source: 4 x b128 loads + packed converts.
static __device__ __forceinline__ v16h load_frag(const float* __restrict__ row, int koff)
{
    const float4* p = (const float4*)(row + koff);
    const float4* q = (const float4*)(row + koff + 16);
    return cvt16(p[0], p[1], q[0], q[1]);
}

// f16 source: 2 x b128 loads, no conversion.
static __device__ __forceinline__ v16h load_frag(const _Float16* __restrict__ row, int koff)
{
    const v8h lo = *(const v8h*)(row + koff);
    const v8h hi = *(const v8h*)(row + koff + 16);
    return __builtin_shufflevector(lo, hi, 0, 1, 2, 3, 4, 5, 6, 7,
                                   8, 9, 10, 11, 12, 13, 14, 15);
}

// fp32 source with elementwise product against qv (for hp_l * hc_c fragments).
static __device__ __forceinline__ v16h load_frag_mul(const float* __restrict__ row,
                                                     const float* __restrict__ qv, int koff)
{
    const float4* p  = (const float4*)(row + koff);
    const float4* q  = (const float4*)(row + koff + 16);
    const float4* pg = (const float4*)(qv + koff);
    const float4* qg = (const float4*)(qv + koff + 16);
    float4 f0 = p[0], f1 = p[1], f2 = q[0], f3 = q[1];
    const float4 g0 = pg[0], g1 = pg[1], g2 = qg[0], g3 = qg[1];
    f0.x *= g0.x; f0.y *= g0.y; f0.z *= g0.z; f0.w *= g0.w;
    f1.x *= g1.x; f1.y *= g1.y; f1.z *= g1.z; f1.w *= g1.w;
    f2.x *= g2.x; f2.y *= g2.y; f2.z *= g2.z; f2.w *= g2.w;
    f3.x *= g3.x; f3.y *= g3.y; f3.z *= g3.z; f3.w *= g3.w;
    return cvt16(f0, f1, f2, f3);
}

// Block-wide softmax over 128 entries (blockDim.x == 256).
#define BLOCK_SOFTMAX_128(scorearr, redarr, tid)                                          \
    do {                                                                                  \
        redarr[tid] = (tid < 128) ? scorearr[tid] : -1e30f;                               \
        __syncthreads();                                                                  \
        for (int s_ = 128; s_ > 0; s_ >>= 1) {                                            \
            if (tid < s_) redarr[tid] = fmaxf(redarr[tid], redarr[tid + s_]);             \
            __syncthreads();                                                              \
        }                                                                                 \
        const float mx_ = redarr[0];                                                      \
        __syncthreads();                                                                  \
        const float e_ = (tid < 128) ? expf(scorearr[tid] - mx_) : 0.f;                   \
        redarr[tid] = e_;                                                                 \
        __syncthreads();                                                                  \
        for (int s_ = 128; s_ > 0; s_ >>= 1) {                                            \
            if (tid < s_) redarr[tid] += redarr[tid + s_];                                 \
            __syncthreads();                                                              \
        }                                                                                 \
        const float inv_ = 1.f / redarr[0];                                               \
        __syncthreads();                                                                  \
        if (tid < 128) scorearr[tid] = e_ * inv_;                                         \
        __syncthreads();                                                                  \
    } while (0)

// ---------------------------------------------------------------------------
// Batched GEMM:  C[M,N] = A[M,K] @ B[N,K]^T (+ bias[N]).
// One wave per block computing a (16*MT) x 64 strip of C: MT*4 accumulators,
// each A fragment feeds 4 v_wmma, each B fragment feeds MT v_wmma.
// Requires M%(16*MT)==0, N%64==0, K%32==0.
// ---------------------------------------------------------------------------
template <typename TA, typename TB, int MT>
__global__ void gemm_bt(const TA* __restrict__ A, const TB* __restrict__ Bw,
                        const float* __restrict__ bias, float* __restrict__ C,
                        int M, int N, int K, int lda, int ldb, int ldc,
                        long long sA, long long sB, long long sC)
{
    (void)M;
    const long long bt = blockIdx.y;
    A  += bt * sA;  Bw += bt * sB;  C += bt * sC;
    const int ngrp = N >> 6;
    const int tm = blockIdx.x / ngrp, tg = blockIdx.x % ngrp;
    const int lane = threadIdx.x & 31;
    const int row0 = tm * (16 * MT), col0 = tg << 6;
    const int kb = (lane >> 4) << 3;

    const TA* arow[MT];
#pragma unroll
    for (int mi = 0; mi < MT; ++mi)
        arow[mi] = A + (size_t)(row0 + mi * 16 + (lane & 15)) * lda;

    v8f acc[MT][4] = {};
    for (int k0 = 0; k0 < K; k0 += 32) {
        v16h a[MT];
#pragma unroll
        for (int mi = 0; mi < MT; ++mi) a[mi] = load_frag(arow[mi], k0 + kb);
#pragma unroll
        for (int j = 0; j < 4; ++j) {
            const int n = col0 + (j << 4) + (lane & 15);
            const v16h b = load_frag(Bw + (size_t)n * ldb, k0 + kb);
#pragma unroll
            for (int mi = 0; mi < MT; ++mi)
                acc[mi][j] = __builtin_amdgcn_wmma_f32_16x16x32_f16(false, a[mi], false, b,
                                                                    (short)0, acc[mi][j],
                                                                    false, false);
        }
    }
#pragma unroll
    for (int mi = 0; mi < MT; ++mi) {
#pragma unroll
        for (int j = 0; j < 4; ++j) {
            const int n = col0 + (j << 4) + (lane & 15);
            const float bv = (bias != nullptr) ? bias[n] : 0.f;
#pragma unroll
            for (int r = 0; r < 8; ++r)
                C[(size_t)(row0 + mi * 16 + r + kb) * ldc + n] = acc[mi][j][r] + bv;
        }
    }
}

// ---------------------------------------------------------------------------
// Elementwise f32 -> f16 conversion (weight staging).
// ---------------------------------------------------------------------------
__global__ void cvt_f16(const float* __restrict__ in, _Float16* __restrict__ out, int n)
{
    const int i = blockIdx.x * blockDim.x + threadIdx.x;
    if (i < n) out[i] = (_Float16)in[i];
}

// Embedding gather, zero-padded to EP_ columns, emitted as f16 (GEMM-only).
__global__ void gather_emb_pad16(const int* __restrict__ idx, const float* __restrict__ emb,
                                 _Float16* __restrict__ x, int rows)
{
    const int gi = blockIdx.x * blockDim.x + threadIdx.x;
    if (gi < rows * EP_) {
        const int r = gi / EP_, e = gi % EP_;
        x[gi] = (e < E_) ? (_Float16)emb[(size_t)idx[r] * E_ + e] : (_Float16)0.f;
    }
}

// Pad a (3H, E) weight to (3H, EP_) with zeros, emitted as f16.
__global__ void pad_wih16(const float* __restrict__ w, _Float16* __restrict__ out)
{
    const int gi = blockIdx.x * blockDim.x + threadIdx.x;
    if (gi < G3_ * EP_) {
        const int r = gi / EP_, e = gi % EP_;
        out[gi] = (e < E_) ? (_Float16)w[(size_t)r * E_ + e] : (_Float16)0.f;
    }
}

// ---------------------------------------------------------------------------
// Persistent GRU recurrence, one block per direction (grid.x = 2).
// xg (B, L, 3H) precomputed (x@Wih^T + bih).  h lives in LDS in both fp32
// (gate math) and f16 (WMMA A fragments, pure ds_load_b128); each step the
// 16x128 @ 128x384 gate matmul runs on WMMA (12 waves x 2 N-tiles, f16
// weights), then 512 threads do the gate nonlinearity.  Backward direction
// indexes the sequence reversed and writes to the original positions.
// ---------------------------------------------------------------------------
__global__ void gru_rec(const float* __restrict__ xg_f, const float* __restrict__ xg_b,
                        const _Float16* __restrict__ Whh_f, const _Float16* __restrict__ Whh_b,
                        const float* __restrict__ bhh_f, const float* __restrict__ bhh_b,
                        float* __restrict__ hout /* (B, L, 2H): fwd cols [0,H), bwd [H,2H) */)
{
    const int dir = blockIdx.x;
    const float* xg  = dir ? xg_b  : xg_f;
    const _Float16* Whh = dir ? Whh_b : Whh_f;
    const float* bhh = dir ? bhh_b : bhh_f;
    const int tid = threadIdx.x;              // 512 threads = 16 waves
    const int lane = tid & 31, wave = tid >> 5;
    const int kb = (lane >> 4) << 3;

    __shared__ float    hbuf[B_][H_];         // hidden state, fp32 (gate math)
    __shared__ _Float16 h16[B_][H_];          // hidden state, f16 (WMMA A side)
    __shared__ float    gh[B_][G3_];          // h @ Whh^T for this step

    for (int i = tid; i < B_ * H_; i += 512) {
        ((float*)hbuf)[i] = 0.f;
        ((_Float16*)h16)[i] = (_Float16)0.f;
    }
    __syncthreads();

    for (int t = 0; t < LP_; ++t) {
        const int xt = dir ? (LP_ - 1 - t) : t;
        if (wave < 12) {                       // 24 N-tiles of 16, 2 per wave
            v8f acc[2] = {};
            const _Float16* hrow = &h16[lane & 15][0];
#pragma unroll
            for (int ks = 0; ks < 4; ++ks) {
                const int koff = (ks << 5) + kb;
                const v16h a = load_frag(hrow, koff);
#pragma unroll
                for (int q = 0; q < 2; ++q) {
                    const int n = ((wave * 2 + q) << 4) + (lane & 15);
                    const v16h b = load_frag(Whh + (size_t)n * H_, koff);
                    acc[q] = __builtin_amdgcn_wmma_f32_16x16x32_f16(false, a, false, b,
                                                                    (short)0, acc[q], false, false);
                }
            }
#pragma unroll
            for (int q = 0; q < 2; ++q) {
                const int n = ((wave * 2 + q) << 4) + (lane & 15);
#pragma unroll
                for (int r = 0; r < 8; ++r) gh[r + kb][n] = acc[q][r];
            }
        }
        __syncthreads();
#pragma unroll
        for (int i = 0; i < 4; ++i) {          // 2048 = B*H gate updates
            const int idx = tid + 512 * i;
            const int b = idx >> 7, j = idx & 127;
            const float* xrow = &xg[((size_t)(b * LP_ + xt)) * G3_];
            const float hr = gh[b][j]          + bhh[j];
            const float hz = gh[b][H_ + j]     + bhh[H_ + j];
            const float hn = gh[b][2 * H_ + j] + bhh[2 * H_ + j];
            const float r  = 1.f / (1.f + expf(-(xrow[j] + hr)));
            const float z  = 1.f / (1.f + expf(-(xrow[H_ + j] + hz)));
            const float nn = tanhf(xrow[2 * H_ + j] + r * hn);
            const float h2 = (1.f - z) * nn + z * hbuf[b][j];
            hbuf[b][j] = h2;
            h16[b][j]  = (_Float16)h2;
            hout[((size_t)(b * LP_ + xt)) * D2_ + dir * H_ + j] = h2;
        }
        __syncthreads();
    }
}

// ---------------------------------------------------------------------------
// Fused elementwise-product attention (dot & self flavors).
// For block (b,c):  S[l,h] = sum_k (P[b,l,k] * Q[b,c,k]) * W[h,k]   (WMMA)
//                   a = softmax_l( v . tanh(S[l,:]) )
//                   out[b,c,:] = sum_l a_l * Vals[b,l,:]
// 256 threads = 8 waves; wave w owns M-tile w with 8 accumulators, so each
// A fragment feeds 8 v_wmma.  W is pre-converted f16.  S lives in dynamic
// LDS with an odd row stride (SLD_=129) so the per-row tanh reduction reads
// are bank-conflict free.
// ---------------------------------------------------------------------------
__global__ void elemattn(const float* __restrict__ P, const float* __restrict__ Q,
                         const _Float16* __restrict__ W, const float* __restrict__ v,
                         const float* __restrict__ Vals, float* __restrict__ out)
{
    extern __shared__ float smem[];
    float* S     = smem;                 // 128*SLD_
    float* qrow  = S + LP_ * SLD_;       // 256
    float* score = qrow + D2_;           // 128
    float* red   = score + LP_;          // 256

    const int b = blockIdx.x >> 7, c = blockIdx.x & 127;
    const int tid = threadIdx.x, lane = tid & 31, wave = tid >> 5;
    const int kb = (lane >> 4) << 3;

    if (tid < D2_) qrow[tid] = Q[((size_t)(b * LC_ + c)) * D2_ + tid];
    __syncthreads();

    const int mt = wave;
    const int m  = (mt << 4) + (lane & 15);
    const float* prow = P + ((size_t)(b * LP_ + m)) * D2_;
    v8f acc[8] = {};
    for (int ks = 0; ks < 8; ++ks) {
        const int koff = (ks << 5) + kb;
        const v16h a = load_frag_mul(prow, qrow, koff);
#pragma unroll
        for (int nt = 0; nt < 8; ++nt) {
            const int n = (nt << 4) + (lane & 15);
            const v16h bb = load_frag(W + (size_t)n * D2_, koff);
            acc[nt] = __builtin_amdgcn_wmma_f32_16x16x32_f16(false, a, false, bb,
                                                             (short)0, acc[nt], false, false);
        }
    }
#pragma unroll
    for (int nt = 0; nt < 8; ++nt) {
        const int n = (nt << 4) + (lane & 15);
#pragma unroll
        for (int r = 0; r < 8; ++r)
            S[((mt << 4) + r + kb) * SLD_ + n] = acc[nt][r];
    }
    __syncthreads();

    if (tid < LP_) {
        float t = 0.f;
        for (int h = 0; h < H_; ++h) t += v[h] * tanhf(S[tid * SLD_ + h]);
        score[tid] = t;
    }
    __syncthreads();
    BLOCK_SOFTMAX_128(score, red, tid);

    float accd = 0.f;
    for (int l = 0; l < LP_; ++l)
        accd += score[l] * Vals[((size_t)(b * LP_ + l)) * D2_ + tid];
    out[((size_t)(b * LC_ + c)) * D2_ + tid] = accd;
}

// ---------------------------------------------------------------------------
// Additive attention: a = softmax_l( v . tanh(P[b,l,:] + sign*Q[b,c,:]) ),
// out[b,c,:] = sum_l a_l Vals[b,l,:].  Q==nullptr -> plain tanh attention (ap).
// Covers ptc (sign=+1), ptm (sign=-1) and ap/rp (C=1, Q=null).
// ---------------------------------------------------------------------------
__global__ void addtanh_attn(const float* __restrict__ P, const float* __restrict__ Q,
                             const float* __restrict__ v, const float* __restrict__ Vals,
                             float* __restrict__ out, float sign, int C)
{
    const int blk = blockIdx.x;
    const int b = blk / C, c = blk % C;
    const int tid = threadIdx.x;  // 256
    __shared__ float qbuf[H_];
    __shared__ float score[LP_];
    __shared__ float red[256];

    if (tid < H_) qbuf[tid] = (Q != nullptr) ? Q[((size_t)(b * C + c)) * H_ + tid] : 0.f;
    __syncthreads();

    if (tid < LP_) {
        const float* prow = P + ((size_t)(b * LP_ + tid)) * H_;
        float t = 0.f;
        for (int h = 0; h < H_; ++h) t += v[h] * tanhf(prow[h] + sign * qbuf[h]);
        score[tid] = t;
    }
    __syncthreads();
    BLOCK_SOFTMAX_128(score, red, tid);

    float accd = 0.f;
    for (int l = 0; l < LP_; ++l)
        accd += score[l] * Vals[((size_t)(b * LP_ + l)) * D2_ + tid];
    out[((size_t)(b * C + c)) * D2_ + tid] = accd;
}

// ---------------------------------------------------------------------------
// Softmax over precomputed scores S (B, LC, LP) + weighted sum of Vals (ptb).
// ---------------------------------------------------------------------------
__global__ void softmax_wsum(const float* __restrict__ S, const float* __restrict__ Vals,
                             float* __restrict__ out)
{
    const int b = blockIdx.x >> 7, c = blockIdx.x & 127;
    const int tid = threadIdx.x;  // 256
    __shared__ float score[LP_];
    __shared__ float red[256];

    if (tid < LP_) score[tid] = S[((size_t)(b * LC_ + c)) * LP_ + tid];
    __syncthreads();
    BLOCK_SOFTMAX_128(score, red, tid);

    float accd = 0.f;
    for (int l = 0; l < LP_; ++l)
        accd += score[l] * Vals[((size_t)(b * LP_ + l)) * D2_ + tid];
    out[((size_t)(b * LC_ + c)) * D2_ + tid] = accd;
}

// ---------------------------------------------------------------------------
// agg_in = concat([hc, pts, ptc, ptd, ptb, ptm], -1) -> f16 (GEMM-only input)
// ---------------------------------------------------------------------------
__global__ void concat6_16(const float* __restrict__ s0, const float* __restrict__ s1,
                           const float* __restrict__ s2, const float* __restrict__ s3,
                           const float* __restrict__ s4, const float* __restrict__ s5,
                           _Float16* __restrict__ out, int total)
{
    const int idx = blockIdx.x * blockDim.x + threadIdx.x;
    if (idx >= total) return;
    const int row = idx / DA_, d = idx % DA_;
    const int seg = d >> 8, off = d & 255;
    const float* s = (seg == 0) ? s0 : (seg == 1) ? s1 : (seg == 2) ? s2
                   : (seg == 3) ? s3 : (seg == 4) ? s4 : s5;
    out[idx] = (_Float16)s[(size_t)row * D2_ + off];
}

// ---------------------------------------------------------------------------
// Final readout: sc[b,c] = vc . (aggWc1[b,c,:] + rpWc2[b,:]); ac = softmax_c;
// rc = sum_c ac * agg[b,c,:];  out[b] = sigmoid(rc . Wpred)
// ---------------------------------------------------------------------------
__global__ void final_readout(const float* __restrict__ aggWc1, const float* __restrict__ rpWc2,
                              const float* __restrict__ vc, const float* __restrict__ agg,
                              const float* __restrict__ Wpred, float* __restrict__ out)
{
    const int b = blockIdx.x;
    const int tid = threadIdx.x;  // 256
    __shared__ float rpw[H_];
    __shared__ float score[LC_];
    __shared__ float red[256];

    if (tid < H_) rpw[tid] = rpWc2[(size_t)b * H_ + tid];
    __syncthreads();
    if (tid < LC_) {
        float t = 0.f;
        for (int h = 0; h < H_; ++h)
            t += vc[h] * (aggWc1[((size_t)(b * LC_ + tid)) * H_ + h] + rpw[h]);
        score[tid] = t;
    }
    __syncthreads();
    BLOCK_SOFTMAX_128(score, red, tid);

    float accd = 0.f;  // tid == output dim d (0..255)
    for (int c = 0; c < LC_; ++c)
        accd += score[c] * agg[((size_t)(b * LC_ + c)) * D2_ + tid];
    red[tid] = accd * Wpred[tid];
    __syncthreads();
    for (int s_ = 128; s_ > 0; s_ >>= 1) {
        if (tid < s_) red[tid] += red[tid + s_];
        __syncthreads();
    }
    if (tid == 0) out[b] = 1.f / (1.f + expf(-red[0]));
}

// ---------------------------------------------------------------------------
// Host side
// ---------------------------------------------------------------------------
template <typename TA, typename TB>
static void gemm(const TA* A, const TB* Bw, const float* bias, float* C,
                 int M, int N, int K, int lda, int ldb, int ldc,
                 long long sA, long long sB, long long sC, int nb, hipStream_t st)
{
    if (M % 32 == 0) {
        dim3 grid((M / 32) * (N / 64), nb);
        gemm_bt<TA, TB, 2><<<grid, 32, 0, st>>>(A, Bw, bias, C, M, N, K, lda, ldb, ldc,
                                                sA, sB, sC);
    } else {
        dim3 grid((M / 16) * (N / 64), nb);
        gemm_bt<TA, TB, 1><<<grid, 32, 0, st>>>(A, Bw, bias, C, M, N, K, lda, ldb, ldc,
                                                sA, sB, sC);
    }
}

static void cvt(const float* in, _Float16* out, int n, hipStream_t st)
{
    cvt_f16<<<(n + 255) / 256, 256, 0, st>>>(in, out, n);
}

extern "C" void kernel_launch(void* const* d_in, const int* in_sizes, int n_in,
                              void* d_out, int out_size, void* d_ws, size_t ws_size,
                              hipStream_t stream)
{
    (void)in_sizes; (void)n_in; (void)out_size; (void)ws_size;

    const int*   post     = (const int*)d_in[0];
    const int*   comm     = (const int*)d_in[1];
    const float* emb      = (const float*)d_in[2];
    const float* pe_Wih_f = (const float*)d_in[3];
    const float* pe_Whh_f = (const float*)d_in[4];
    const float* pe_bih_f = (const float*)d_in[5];
    const float* pe_bhh_f = (const float*)d_in[6];
    const float* pe_Wih_b = (const float*)d_in[7];
    const float* pe_Whh_b = (const float*)d_in[8];
    const float* pe_bih_b = (const float*)d_in[9];
    const float* pe_bhh_b = (const float*)d_in[10];
    const float* ce_Wih_f = (const float*)d_in[11];
    const float* ce_Whh_f = (const float*)d_in[12];
    const float* ce_bih_f = (const float*)d_in[13];
    const float* ce_bhh_f = (const float*)d_in[14];
    const float* ce_Wih_b = (const float*)d_in[15];
    const float* ce_Whh_b = (const float*)d_in[16];
    const float* ce_bih_b = (const float*)d_in[17];
    const float* ce_bhh_b = (const float*)d_in[18];
    const float* ag_Wih_f = (const float*)d_in[19];
    const float* ag_Whh_f = (const float*)d_in[20];
    const float* ag_bih_f = (const float*)d_in[21];
    const float* ag_bhh_f = (const float*)d_in[22];
    const float* ag_Wih_b = (const float*)d_in[23];
    const float* ag_Whh_b = (const float*)d_in[24];
    const float* ag_bih_b = (const float*)d_in[25];
    const float* ag_bhh_b = (const float*)d_in[26];
    const float* Wc1   = (const float*)d_in[27];
    const float* Wc2   = (const float*)d_in[28];
    const float* vc    = (const float*)d_in[29];
    const float* Wb    = (const float*)d_in[30];
    const float* Wd    = (const float*)d_in[31];
    const float* vd    = (const float*)d_in[32];
    const float* Wm    = (const float*)d_in[33];
    const float* vm    = (const float*)d_in[34];
    const float* Ws    = (const float*)d_in[35];
    const float* vs    = (const float*)d_in[36];
    const float* Wp    = (const float*)d_in[37];
    const float* vp    = (const float*)d_in[38];
    const float* Wpred = (const float*)d_in[39];

    const int R = B_ * LP_;  // 2048 rows for both sequences

    // Workspace partitioning (byte-based, 64B-aligned slots)
    char* base = (char*)d_ws;
    size_t off = 0;
    auto allocf = [&](size_t n) { float* p = (float*)(base + off);
                                  off += ((n * 4 + 63) & ~(size_t)63); return p; };
    auto alloch = [&](size_t n) { _Float16* p = (_Float16*)(base + off);
                                  off += ((n * 2 + 63) & ~(size_t)63); return p; };

    // fp32 intermediates
    float* xg_pe_f = allocf((size_t)R * G3_);
    float* xg_pe_b = allocf((size_t)R * G3_);
    float* xg_ce_f = allocf((size_t)R * G3_);
    float* xg_ce_b = allocf((size_t)R * G3_);
    float* xg_ag_f = allocf((size_t)R * G3_);
    float* xg_ag_b = allocf((size_t)R * G3_);
    float* hp      = allocf((size_t)R * D2_);
    float* hc      = allocf((size_t)R * D2_);
    float* hpWc1   = allocf((size_t)R * H_);
    float* hcWc2   = allocf((size_t)R * H_);
    float* hpWm    = allocf((size_t)R * H_);
    float* hcWm    = allocf((size_t)R * H_);
    float* hpWb    = allocf((size_t)R * D2_);
    float* hpWp    = allocf((size_t)R * H_);
    float* Sb      = allocf((size_t)B_ * LC_ * LP_);
    float* ptc     = allocf((size_t)R * D2_);
    float* ptb     = allocf((size_t)R * D2_);
    float* ptd     = allocf((size_t)R * D2_);
    float* ptm     = allocf((size_t)R * D2_);
    float* pts     = allocf((size_t)R * D2_);
    float* aggout  = allocf((size_t)R * D2_);
    float* aggWc1  = allocf((size_t)R * H_);
    float* rp      = allocf((size_t)B_ * D2_);
    float* rpWc2   = allocf((size_t)B_ * H_);

    // f16 staging: GEMM-only activations + all weight operands
    _Float16* xpost16   = alloch((size_t)R * EP_);
    _Float16* xcomm16   = alloch((size_t)R * EP_);
    _Float16* agg_in16  = alloch((size_t)R * DA_);
    _Float16* hpWb16    = alloch((size_t)R * D2_);
    _Float16* wih_pe_f  = alloch((size_t)G3_ * EP_);
    _Float16* wih_pe_b  = alloch((size_t)G3_ * EP_);
    _Float16* wih_ce_f  = alloch((size_t)G3_ * EP_);
    _Float16* wih_ce_b  = alloch((size_t)G3_ * EP_);
    _Float16* wih_ag_f  = alloch((size_t)G3_ * DA_);
    _Float16* wih_ag_b  = alloch((size_t)G3_ * DA_);
    _Float16* whh_pe_f  = alloch((size_t)G3_ * H_);
    _Float16* whh_pe_b  = alloch((size_t)G3_ * H_);
    _Float16* whh_ce_f  = alloch((size_t)G3_ * H_);
    _Float16* whh_ce_b  = alloch((size_t)G3_ * H_);
    _Float16* whh_ag_f  = alloch((size_t)G3_ * H_);
    _Float16* whh_ag_b  = alloch((size_t)G3_ * H_);
    _Float16* Wc1_16    = alloch((size_t)H_ * D2_);
    _Float16* Wc2_16    = alloch((size_t)H_ * D2_);
    _Float16* Wm_16     = alloch((size_t)H_ * D2_);
    _Float16* Wb_16     = alloch((size_t)D2_ * D2_);
    _Float16* Wp_16     = alloch((size_t)H_ * D2_);
    _Float16* Wd_16     = alloch((size_t)H_ * D2_);
    _Float16* Ws_16     = alloch((size_t)H_ * D2_);

    // 0) Weight staging: pad + convert to f16
    const int wpad_blocks = (G3_ * EP_ + 255) / 256;
    pad_wih16<<<wpad_blocks, 256, 0, stream>>>(pe_Wih_f, wih_pe_f);
    pad_wih16<<<wpad_blocks, 256, 0, stream>>>(pe_Wih_b, wih_pe_b);
    pad_wih16<<<wpad_blocks, 256, 0, stream>>>(ce_Wih_f, wih_ce_f);
    pad_wih16<<<wpad_blocks, 256, 0, stream>>>(ce_Wih_b, wih_ce_b);
    cvt(ag_Wih_f, wih_ag_f, G3_ * DA_, stream);
    cvt(ag_Wih_b, wih_ag_b, G3_ * DA_, stream);
    cvt(pe_Whh_f, whh_pe_f, G3_ * H_, stream);
    cvt(pe_Whh_b, whh_pe_b, G3_ * H_, stream);
    cvt(ce_Whh_f, whh_ce_f, G3_ * H_, stream);
    cvt(ce_Whh_b, whh_ce_b, G3_ * H_, stream);
    cvt(ag_Whh_f, whh_ag_f, G3_ * H_, stream);
    cvt(ag_Whh_b, whh_ag_b, G3_ * H_, stream);
    cvt(Wc1, Wc1_16, H_ * D2_, stream);
    cvt(Wc2, Wc2_16, H_ * D2_, stream);
    cvt(Wm,  Wm_16,  H_ * D2_, stream);
    cvt(Wb,  Wb_16,  D2_ * D2_, stream);
    cvt(Wp,  Wp_16,  H_ * D2_, stream);
    cvt(Wd,  Wd_16,  H_ * D2_, stream);
    cvt(Ws,  Ws_16,  H_ * D2_, stream);

    // 1) Embedding gathers (zero-padded to EP_, f16)
    gather_emb_pad16<<<(R * EP_ + 255) / 256, 256, 0, stream>>>(post, emb, xpost16, R);
    gather_emb_pad16<<<(R * EP_ + 255) / 256, 256, 0, stream>>>(comm, emb, xcomm16, R);

    // 2) GRU input projections (x @ Wih^T + bih), K padded to 320, f16 x f16
    gemm(xpost16, wih_pe_f, pe_bih_f, xg_pe_f, R, G3_, EP_, EP_, EP_, G3_, 0, 0, 0, 1, stream);
    gemm(xpost16, wih_pe_b, pe_bih_b, xg_pe_b, R, G3_, EP_, EP_, EP_, G3_, 0, 0, 0, 1, stream);
    gemm(xcomm16, wih_ce_f, ce_bih_f, xg_ce_f, R, G3_, EP_, EP_, EP_, G3_, 0, 0, 0, 1, stream);
    gemm(xcomm16, wih_ce_b, ce_bih_b, xg_ce_b, R, G3_, EP_, EP_, EP_, G3_, 0, 0, 0, 1, stream);

    // 3) BiGRU recurrences -> hp, hc (f16 weights)
    gru_rec<<<2, 512, 0, stream>>>(xg_pe_f, xg_pe_b, whh_pe_f, whh_pe_b, pe_bhh_f, pe_bhh_b, hp);
    gru_rec<<<2, 512, 0, stream>>>(xg_ce_f, xg_ce_b, whh_ce_f, whh_ce_b, ce_bhh_f, ce_bhh_b, hc);

    // 4) Shared projections (f32 activations x f16 weights)
    gemm(hp, Wc1_16, nullptr, hpWc1, R, H_,  D2_, D2_, D2_, H_,  0, 0, 0, 1, stream);
    gemm(hc, Wc2_16, nullptr, hcWc2, R, H_,  D2_, D2_, D2_, H_,  0, 0, 0, 1, stream);
    gemm(hp, Wm_16,  nullptr, hpWm,  R, H_,  D2_, D2_, D2_, H_,  0, 0, 0, 1, stream);
    gemm(hc, Wm_16,  nullptr, hcWm,  R, H_,  D2_, D2_, D2_, H_,  0, 0, 0, 1, stream);
    gemm(hp, Wb_16,  nullptr, hpWb,  R, D2_, D2_, D2_, D2_, D2_, 0, 0, 0, 1, stream);
    gemm(hp, Wp_16,  nullptr, hpWp,  R, H_,  D2_, D2_, D2_, H_,  0, 0, 0, 1, stream);

    // 5) Attention mechanisms
    addtanh_attn<<<B_ * LC_, 256, 0, stream>>>(hpWc1, hcWc2, vc, hp, ptc,  1.f, LC_);
    addtanh_attn<<<B_ * LC_, 256, 0, stream>>>(hpWm,  hcWm,  vm, hp, ptm, -1.f, LC_);

    // bilinear: Sb[b] = hc[b] @ hpWb[b]^T (B side staged f16), then softmax+sum
    cvt(hpWb, hpWb16, R * D2_, stream);
    gemm(hc, hpWb16, nullptr, Sb, LC_, LP_, D2_, D2_, D2_, LP_,
         (long long)LC_ * D2_, (long long)LP_ * D2_, (long long)LC_ * LP_, B_, stream);
    softmax_wsum<<<B_ * LC_, 256, 0, stream>>>(Sb, hp, ptb);

    // dot & self attention (fused WMMA), dynamic LDS = S(128*129) + qrow + score + red
    const size_t ea_lds = (size_t)(LP_ * SLD_ + D2_ + LP_ + 256) * sizeof(float);
    elemattn<<<B_ * LC_, 256, ea_lds, stream>>>(hp, hc, Wd_16, vd, hp, ptd);
    elemattn<<<B_ * LC_, 256, ea_lds, stream>>>(hc, hc, Ws_16, vs, hc, pts);

    // 6) Aggregation BiGRU (agg_in staged directly as f16)
    concat6_16<<<(R * DA_ + 255) / 256, 256, 0, stream>>>(hc, pts, ptc, ptd, ptb, ptm,
                                                          agg_in16, R * DA_);
    gemm(agg_in16, wih_ag_f, ag_bih_f, xg_ag_f, R, G3_, DA_, DA_, DA_, G3_, 0, 0, 0, 1, stream);
    gemm(agg_in16, wih_ag_b, ag_bih_b, xg_ag_b, R, G3_, DA_, DA_, DA_, G3_, 0, 0, 0, 1, stream);
    gru_rec<<<2, 512, 0, stream>>>(xg_ag_f, xg_ag_b, whh_ag_f, whh_ag_b, ag_bhh_f, ag_bhh_b, aggout);

    // 7) Readout
    gemm(aggout, Wc1_16, nullptr, aggWc1, R,  H_, D2_, D2_, D2_, H_, 0, 0, 0, 1, stream);
    addtanh_attn<<<B_, 256, 0, stream>>>(hpWp, nullptr, vp, hp, rp, 1.f, 1);
    gemm(rp, Wc2_16, nullptr, rpWc2, B_, H_, D2_, D2_, D2_, H_, 0, 0, 0, 1, stream);
    final_readout<<<B_, 256, 0, stream>>>(aggWc1, rpWc2, vc, aggout, Wpred, (float*)d_out);
}
